// PointSetPooling_17841294148335
// MI455X (gfx1250) — matrix operand order repaired and checked by reading, not compiled, not run on hardware
//
#include <hip/hip_runtime.h>
#include <hip/hip_bf16.h>

typedef __attribute__((ext_vector_type(16))) __bf16 v16bf;
typedef __attribute__((ext_vector_type(8)))  float  v8f;

#define EPS 1e-5f

// ---------------------------------------------------------------------------
// Build X0[s, 0:4] = [feat(pt), coords(pt) - coords(kp(seg))], K padded to 32.
// ---------------------------------------------------------------------------
__global__ void psp_gather(const float* __restrict__ pf, const float* __restrict__ pc,
                           const int* __restrict__ kpi, const int* __restrict__ si,
                           float* __restrict__ X0, int S) {
    int s = blockIdx.x * blockDim.x + threadIdx.x;
    if (s >= S) return;
    int p  = si[2 * s + 0];
    int g  = si[2 * s + 1];
    int kp = kpi[g];
    float* row = X0 + (size_t)s * 32;
    row[0] = pf[p];
    row[1] = pc[3 * p + 0] - pc[3 * kp + 0];
    row[2] = pc[3 * p + 1] - pc[3 * kp + 1];
    row[3] = pc[3 * p + 2] - pc[3 * kp + 2];
#pragma unroll
    for (int k = 4; k < 32; ++k) row[k] = 0.0f;
}

__global__ void psp_fill_zero(float* __restrict__ p, size_t n) {
    size_t i = blockIdx.x * (size_t)blockDim.x + threadIdx.x;
    size_t stride = (size_t)gridDim.x * blockDim.x;
    for (; i < n; i += stride) p[i] = 0.0f;
}

// ---------------------------------------------------------------------------
// One-shot per layer: transpose + zero-pad + f32->bf16 convert the weights.
// Wt[c][k] for c in [0,Cpad), k in [0,ldK). Also pads the bias.
// ---------------------------------------------------------------------------
__global__ void psp_prep_w(const float* __restrict__ W, const float* __restrict__ bias,
                           __bf16* __restrict__ Wt, float* __restrict__ bpad,
                           int Cin, int Cout, int ldK, int Cpad) {
    int idx = blockIdx.x * blockDim.x + threadIdx.x;
    int total = Cpad * ldK;
    if (idx < total) {
        int c = idx / ldK;
        int k = idx - c * ldK;
        float v = (k < Cin && c < Cout) ? W[(size_t)k * Cout + c] : 0.0f;
        Wt[(size_t)c * ldK + k] = (__bf16)v;
    }
    if (idx < Cpad) bpad[idx] = (idx < Cout) ? bias[idx] : 0.0f;
}

// ---------------------------------------------------------------------------
// GEMM: Y[M, ldY] = relu(X[M, ldX] @ W + b) with per-column sum/sumsq stats.
// Wave = 16x(16*NT) D tile. Software pipeline with PEELED last iteration:
// loads in the steady-state loop are unconditional, so the consume point has
// one predecessor and the compiler can use partial s_wait_loadcnt (latency
// hidden under the WMMAs) and unroll/rename away the register rotation.
// ---------------------------------------------------------------------------
template <int NT>
__global__ __launch_bounds__(256) void psp_gemm_bias_relu_stats(
    const float* __restrict__ X, int ldX,
    const __bf16* __restrict__ Wt,      // [Cpad][ldX] bf16, zero-padded
    const float* __restrict__ bpad,     // [Cpad]
    float* __restrict__ Y, int ldY,
    int M, int Cout,
    float* __restrict__ ssum, float* __restrict__ ssq)
{
    const int tid  = threadIdx.x;
    const int wave = tid >> 5;
    const int lane = tid & 31;
    const int half = lane >> 4;
    const int lm   = lane & 15;

    const int rowTile = blockIdx.x * 128 + wave * 16;
    const int colGrp  = blockIdx.y * (16 * NT);

    // Running pointers: A = lane's row (one 16-float run per k-step),
    // B = lane's column of transposed bf16 weights (32B per k-step).
    const float4* xp = (const float4*)(X + (size_t)(rowTile + lm) * ldX) + half * 4;
    const v16bf* bp[NT];
#pragma unroll
    for (int t = 0; t < NT; ++t)
        bp[t] = (const v16bf*)(Wt + (size_t)(colGrp + t * 16 + lm) * ldX) + half;

    v8f acc[NT];
#pragma unroll
    for (int t = 0; t < NT; ++t) acc[t] = (v8f){};

    // ---- prologue: fragment loads for k0 = 0
    float4 a0 = xp[0], a1 = xp[1], a2 = xp[2], a3 = xp[3];
    v16bf b[NT];
#pragma unroll
    for (int t = 0; t < NT; ++t) b[t] = bp[t][0];

    // ---- steady state: unconditional prefetch of next set, consume current
    for (int k0 = 0; k0 + 32 < ldX; k0 += 32) {
        xp += 8;
        float4 na0 = xp[0], na1 = xp[1], na2 = xp[2], na3 = xp[3];
        v16bf nb[NT];
#pragma unroll
        for (int t = 0; t < NT; ++t) { bp[t] += 2; nb[t] = bp[t][0]; }

        v16bf a;
        a[0]  = (__bf16)a0.x; a[1]  = (__bf16)a0.y; a[2]  = (__bf16)a0.z; a[3]  = (__bf16)a0.w;
        a[4]  = (__bf16)a1.x; a[5]  = (__bf16)a1.y; a[6]  = (__bf16)a1.z; a[7]  = (__bf16)a1.w;
        a[8]  = (__bf16)a2.x; a[9]  = (__bf16)a2.y; a[10] = (__bf16)a2.z; a[11] = (__bf16)a2.w;
        a[12] = (__bf16)a3.x; a[13] = (__bf16)a3.y; a[14] = (__bf16)a3.z; a[15] = (__bf16)a3.w;
#pragma unroll
        for (int t = 0; t < NT; ++t)
            acc[t] = __builtin_amdgcn_wmma_f32_16x16x32_bf16(
                false, a, false, b[t], (short)0, acc[t], false, false);

        a0 = na0; a1 = na1; a2 = na2; a3 = na3;
#pragma unroll
        for (int t = 0; t < NT; ++t) b[t] = nb[t];
    }

    // ---- epilogue: consume last fragment set
    {
        v16bf a;
        a[0]  = (__bf16)a0.x; a[1]  = (__bf16)a0.y; a[2]  = (__bf16)a0.z; a[3]  = (__bf16)a0.w;
        a[4]  = (__bf16)a1.x; a[5]  = (__bf16)a1.y; a[6]  = (__bf16)a1.z; a[7]  = (__bf16)a1.w;
        a[8]  = (__bf16)a2.x; a[9]  = (__bf16)a2.y; a[10] = (__bf16)a2.z; a[11] = (__bf16)a2.w;
        a[12] = (__bf16)a3.x; a[13] = (__bf16)a3.y; a[14] = (__bf16)a3.z; a[15] = (__bf16)a3.w;
#pragma unroll
        for (int t = 0; t < NT; ++t)
            acc[t] = __builtin_amdgcn_wmma_f32_16x16x32_bf16(
                false, a, false, b[t], (short)0, acc[t], false, false);
    }

    __shared__ float red[2][8][16 * NT];
#pragma unroll
    for (int t = 0; t < NT; ++t) {
        int col = colGrp + t * 16 + lm;
        const bool colStat = (col < Cout);
        float bcol = bpad[col];                    // padded: always in-bounds
        float s = 0.0f, s2 = 0.0f;
#pragma unroll
        for (int v = 0; v < 8; ++v) {
            int row = rowTile + v + half * 8;      // C/D: VGPR v -> M = v + 8*half
            float val = fmaxf(acc[t][v] + bcol, 0.0f);
            if (!colStat) val = 0.0f;              // keep padding columns zero
            if (col < ldY) Y[(size_t)row * ldY + col] = val;
            s  += val;
            s2 += val * val;
        }
        s  += __shfl_xor(s, 16, 32);
        s2 += __shfl_xor(s2, 16, 32);
        if (half == 0) { red[0][wave][t * 16 + lm] = s; red[1][wave][t * 16 + lm] = s2; }
    }
    __syncthreads();
    if (tid < 32 * NT) {
        int which = (tid >= 16 * NT) ? 1 : 0;
        int c = which ? (tid - 16 * NT) : tid;
        float t = 0.0f;
#pragma unroll
        for (int w = 0; w < 8; ++w) t += red[which][w][c];
        if (colGrp + c < Cout)
            atomicAdd((which ? ssq : ssum) + colGrp + c, t);
    }
}

// ---------------------------------------------------------------------------
__global__ void psp_finalize_stats(const float* __restrict__ ssum, const float* __restrict__ ssq,
                                   const float* __restrict__ g, const float* __restrict__ beta,
                                   float* __restrict__ scale, float* __restrict__ shift,
                                   int C, int Cpad, float invM) {
    int c = threadIdx.x;
    if (c >= Cpad) return;
    if (c < C) {
        float mu  = ssum[c] * invM;
        float var = ssq[c] * invM - mu * mu;
        float inv = rsqrtf(var + EPS);
        float sc  = g[c] * inv;
        scale[c] = sc;
        shift[c] = beta[c] - mu * sc;
    } else {
        scale[c] = 0.0f;
        shift[c] = 0.0f;
    }
}

// One block per row, thread per column (coalesced; no div/mod).
__global__ void psp_apply_norm(float* __restrict__ Y, const float* __restrict__ scale,
                               const float* __restrict__ shift, int ldY, int C) {
    int c = threadIdx.x;
    if (c >= C) return;
    size_t i = (size_t)blockIdx.x * ldY + c;
    Y[i] = Y[i] * scale[c] + shift[c];
}

// ---------------------------------------------------------------------------
// segment_max clamped at 0: buffer pre-zeroed, positive floats order as ints.
// One block per source row, thread per column.
// ---------------------------------------------------------------------------
__global__ void psp_segmax(const float* __restrict__ Y, int ldY, const int* __restrict__ si,
                           float* __restrict__ seg, int ldS, int C) {
    int s = blockIdx.x;
    int c = threadIdx.x;
    if (c >= C) return;
    int gidx = si[2 * s + 1];
    float v = Y[(size_t)s * ldY + c];
    if (v > 0.0f)
        atomicMax((int*)(seg + (size_t)gidx * ldS + c), __float_as_int(v));
}

// ---------------------------------------------------------------------------
extern "C" void kernel_launch(void* const* d_in, const int* in_sizes, int n_in,
                              void* d_out, int out_size, void* d_ws, size_t ws_size,
                              hipStream_t stream) {
    const int K = 16384, S = 262144;

    const float* pf  = (const float*)d_in[0];
    const float* pc  = (const float*)d_in[1];
    const int*   kpi = (const int*)d_in[2];
    const int*   si  = (const int*)d_in[3];

    // params pytree order: sorted dict keys puts 'out' first; detect to be safe.
    const float* ptP[16];
    const float* outP[8];
    if (in_sizes[4] == 300 * 300) {
        for (int i = 0; i < 8; ++i)  outP[i] = (const float*)d_in[4 + i];
        for (int i = 0; i < 16; ++i) ptP[i]  = (const float*)d_in[12 + i];
    } else {
        for (int i = 0; i < 16; ++i) ptP[i]  = (const float*)d_in[4 + i];
        for (int i = 0; i < 8; ++i)  outP[i] = (const float*)d_in[20 + i];
    }

    float* ws = (float*)d_ws;
    size_t off = 0;
    float* X0  = ws + off; off += (size_t)S * 32;
    float* Y1  = ws + off; off += (size_t)S * 32;
    float* Y2  = ws + off; off += (size_t)S * 64;
    float* Y3  = ws + off; off += (size_t)S * 128;
    float* Y4  = ws + off; off += (size_t)S * 320;
    float* SEG = ws + off; off += (size_t)K * 320;
    float* Z1  = ws + off; off += (size_t)K * 320;
    __bf16* WT = (__bf16*)(ws + off); off += 320 * 320 / 2;  // bf16 scratch
    float* bpad  = ws + off; off += 320;
    float* ssum  = ws + off; off += 320;
    float* ssq   = ws + off; off += 320;
    float* scale = ws + off; off += 320;
    float* shift = ws + off; off += 320;

    auto run_layer = [&](const float* X, int ldX, const float* Wm, const float* b,
                         const float* g, const float* beta, float* Y, int ldY,
                         int M, int Cin, int Cout, int NT) {
        int span  = 16 * NT;
        int gridY = (ldY + span - 1) / span;
        int Cpad  = gridY * span;
        psp_fill_zero<<<1, 256, 0, stream>>>(ssum, 640);           // ssum+ssq contiguous
        int totalW = Cpad * ldX;
        psp_prep_w<<<(totalW + 255) / 256, 256, 0, stream>>>(Wm, b, WT, bpad,
                                                             Cin, Cout, ldX, Cpad);
        dim3 grid(M / 128, gridY);
        if (NT == 2)
            psp_gemm_bias_relu_stats<2><<<grid, 256, 0, stream>>>(
                X, ldX, WT, bpad, Y, ldY, M, Cout, ssum, ssq);
        else
            psp_gemm_bias_relu_stats<4><<<grid, 256, 0, stream>>>(
                X, ldX, WT, bpad, Y, ldY, M, Cout, ssum, ssq);
        psp_finalize_stats<<<1, 512, 0, stream>>>(ssum, ssq, g, beta, scale, shift,
                                                  Cout, Cpad, 1.0f / (float)M);
        int nt = (ldY + 31) & ~31;
        psp_apply_norm<<<M, nt, 0, stream>>>(Y, scale, shift, ldY, ldY);
    };

    // 1) gather + concat (K padded 4 -> 32)
    psp_gather<<<(S + 255) / 256, 256, 0, stream>>>(pf, pc, kpi, si, X0, S);

    // 2) point MLP: 4 -> 32 -> 64 -> 128 -> 300 (stored stride 320)
    run_layer(X0, 32,  ptP[0],  ptP[1],  ptP[2],  ptP[3],  Y1, 32,  S, 4,   32,  2);
    run_layer(Y1, 32,  ptP[4],  ptP[5],  ptP[6],  ptP[7],  Y2, 64,  S, 32,  64,  4);
    run_layer(Y2, 64,  ptP[8],  ptP[9],  ptP[10], ptP[11], Y3, 128, S, 64,  128, 4);
    run_layer(Y3, 128, ptP[12], ptP[13], ptP[14], ptP[15], Y4, 320, S, 128, 300, 4);

    // 3) segment_max (clamped at 0 by construction); SEG padding stays zero
    psp_fill_zero<<<2048, 256, 0, stream>>>(SEG, (size_t)K * 320);
    psp_segmax<<<S, 320, 0, stream>>>(Y4, 320, si, SEG, 320, 300);

    // 4) out MLP: 300 -> 300 -> 300; final layer dense into d_out
    run_layer(SEG, 320, outP[0], outP[1], outP[2], outP[3], Z1, 320, K, 300, 300, 4);
    run_layer(Z1,  320, outP[4], outP[5], outP[6], outP[7], (float*)d_out, 300, K, 300, 300, 4);
}
